// AttentionGCN_841813590592
// MI455X (gfx1250) — compile-verified, matrix operand order checked by reading, not run on hardware
//
#include <hip/hip_runtime.h>
#include <hip/hip_bf16.h>
#include <math.h>

// ---------------- problem constants ----------------
#define BB   32
#define NN   1024
#define FF   64
#define PRE  64
#define L1   128

typedef __attribute__((ext_vector_type(16))) _Float16 v16h;
typedef __attribute__((ext_vector_type(8)))  _Float16 v8h;
typedef __attribute__((ext_vector_type(8)))  float    v8f;
typedef __attribute__((ext_vector_type(4)))  float    v4f;

// ---------------------------------------------------------------------------
// Fragment builders for V_WMMA_F32_16X16X32_F16 (wave32).
// 16-bit A 16x32 layout (ISA 7.12.2): lane L (L<16) holds row M=L,
//   halves 0..7 = K {0..7}, halves 8..15 = K {16..23};
//   lane L+16 holds row M=L, K {8..15} and {24..31}.
// B 32x16 is the mirrored layout over N=lane, i.e. the A-layout of B^T,
// so both operands are loaded from K-contiguous ("[row][K]") storage with
// the same per-lane pattern: p = &src[row*ld + kc + kb], kb = (lane>=16)*8.
// ---------------------------------------------------------------------------
__device__ __forceinline__ v16h fragF32(const float* p) {
  v4f a0 = *(const v4f*)(p + 0);
  v4f a1 = *(const v4f*)(p + 4);
  v4f b0 = *(const v4f*)(p + 16);
  v4f b1 = *(const v4f*)(p + 20);
  v16h r;
#pragma unroll
  for (int t = 0; t < 4; ++t) {
    r[t]      = (_Float16)a0[t];
    r[4 + t]  = (_Float16)a1[t];
    r[8 + t]  = (_Float16)b0[t];
    r[12 + t] = (_Float16)b1[t];
  }
  return r;
}

__device__ __forceinline__ v16h fragF16(const _Float16* p) {
  v8h lo = *(const v8h*)(p);        // 16B load (global_load_b128 / ds_load_b128)
  v8h hi = *(const v8h*)(p + 16);
  v16h r;
#pragma unroll
  for (int t = 0; t < 8; ++t) { r[t] = lo[t]; r[8 + t] = hi[t]; }
  return r;
}

// ---------------------------------------------------------------------------
// Kernel 1: deg = sum_j A[b][i][j];  d_isqrt = rsqrt(deg ? deg : 1)
// One wave per row; 8 rows / 256-thread block.
// ---------------------------------------------------------------------------
__global__ void k_degree(const float* __restrict__ A, float* __restrict__ disq) {
  int row  = (blockIdx.x * blockDim.x + threadIdx.x) >> 5;   // 0 .. B*N-1
  int lane = threadIdx.x & 31;
  if (row >= BB * NN) return;
  const float* p = A + (size_t)row * NN;
  float s = 0.f;
  for (int j = lane; j < NN; j += 32) s += p[j];
#pragma unroll
  for (int o = 16; o; o >>= 1) s += __shfl_xor(s, o, 32);
  if (lane == 0) {
    float d = (s != 0.f) ? s : 1.f;
    disq[row] = rsqrtf(d);
  }
}

// ---------------------------------------------------------------------------
// Kernel 2: xsT[b][f][j] = (f16)( d_isqrt[b][j] * x[b][j][f] )
// K-contiguous layout so the GEMM B-fragments are 16-byte loads.
// ---------------------------------------------------------------------------
__global__ void k_xsT(const float* __restrict__ x, const float* __restrict__ disq,
                      _Float16* __restrict__ xsT) {
  int idx = blockIdx.x * 256 + threadIdx.x;     // over B*F*N, order [b][f][j]
  int j = idx & (NN - 1);
  int f = (idx >> 10) & (FF - 1);
  int b = idx >> 16;
  float v = disq[b * NN + j] * x[((size_t)b * NN + j) * FF + f];
  xsT[idx] = (_Float16)v;
}

// ---------------------------------------------------------------------------
// Kernel 3 (fused): per 16-row tile (one wave):
//   h      = alpha*x + d_isqrt(i) * (A(i,:) @ xsT)          128 WMMAs
//   values = relu(h @ W_pre^T + b_pre)   -> global f16        8 WMMAs
//   queries= relu(values @ W_fc1^T + b_fc1)                  16 WMMAs
//   logits = queries @ W_keys^T + b_keys -> global f32  (shfl reduction)
// C-layout -> A-layout reshape goes through a per-wave LDS slice; same-wave
// DS ops are in-order so no workgroup barrier is required.
// ---------------------------------------------------------------------------
__global__ __launch_bounds__(128) void k_fused(
    const float* __restrict__ A, const float* __restrict__ x,
    const float* __restrict__ alpha, const float* __restrict__ disq,
    const _Float16* __restrict__ xsT,
    const float* __restrict__ W_pre, const float* __restrict__ b_pre,
    const float* __restrict__ W_fc1, const float* __restrict__ b_fc1,
    const float* __restrict__ W_keys, const float* __restrict__ b_keys,
    _Float16* __restrict__ values, float* __restrict__ logits) {
  __shared__ _Float16 lds_h[4][16][72];   // 72 = 64 + pad (bank spread)
  __shared__ _Float16 lds_v[4][16][72];

  const int wv     = threadIdx.x >> 5;
  const int lane   = threadIdx.x & 31;
  const int lane15 = lane & 15;
  const int hi8    = (lane >> 4) * 8;     // 0 or 8
  const int kb     = (lane >> 4) * 8;     // K sub-base per half-wave

  const int b  = blockIdx.x >> 4;                       // 32 batches
  const int i0 = (blockIdx.x & 15) * 64 + wv * 16;      // row-tile base

  // ---- stage 1: h = alpha*x + ri * (A @ xsT) --------------------------------
  const float*    Arow  = A   + ((size_t)b * NN + i0 + lane15) * NN + kb;
  const _Float16* xTb   = xsT + (size_t)b * FF * NN;

  v8f acc[4] = {};
  for (int kc = 0; kc < NN; kc += 32) {
    v16h af = fragF32(Arow + kc);
    __builtin_prefetch(Arow + kc + 64, 0, 0);   // A chunk two iterations ahead -> global_prefetch_b8
#pragma unroll
    for (int nt = 0; nt < 4; ++nt) {
      const _Float16* bp = xTb + (size_t)(nt * 16 + lane15) * NN + kc + kb;
      v16h bf = fragF16(bp);
      acc[nt] = __builtin_amdgcn_wmma_f32_16x16x32_f16(
          false, af, false, bf, (short)0, acc[nt], false, false);
    }
  }

  const float alphav = alpha[0];
#pragma unroll
  for (int nt = 0; nt < 4; ++nt) {
    const int col = nt * 16 + lane15;
#pragma unroll
    for (int r = 0; r < 8; ++r) {
      const int row = i0 + hi8 + r;
      float ri = disq[b * NN + row];
      float h  = alphav * x[((size_t)b * NN + row) * FF + col] + ri * acc[nt][r];
      lds_h[wv][hi8 + r][col] = (_Float16)h;
    }
  }

  // ---- stage 2: values = relu(h @ W_pre^T + b_pre) --------------------------
  v8f acc2[4] = {};
#pragma unroll
  for (int kc = 0; kc < PRE; kc += 32) {
    v16h af = fragF16(&lds_h[wv][lane15][kc + kb]);
#pragma unroll
    for (int nt = 0; nt < 4; ++nt) {
      v16h bf = fragF32(W_pre + (nt * 16 + lane15) * PRE + kc + kb);
      acc2[nt] = __builtin_amdgcn_wmma_f32_16x16x32_f16(
          false, af, false, bf, (short)0, acc2[nt], false, false);
    }
  }
#pragma unroll
  for (int nt = 0; nt < 4; ++nt) {
    const int col = nt * 16 + lane15;
    const float bias = b_pre[col];
#pragma unroll
    for (int r = 0; r < 8; ++r) {
      const int row = i0 + hi8 + r;
      float v = fmaxf(acc2[nt][r] + bias, 0.f);
      _Float16 vh = (_Float16)v;
      values[((size_t)b * NN + row) * PRE + col] = vh;
      lds_v[wv][hi8 + r][col] = vh;
    }
  }

  // ---- stage 3: queries = relu(values @ W_fc1^T + b_fc1) --------------------
  v8f acc3[8] = {};
#pragma unroll
  for (int kc = 0; kc < PRE; kc += 32) {
    v16h af = fragF16(&lds_v[wv][lane15][kc + kb]);
#pragma unroll
    for (int nt = 0; nt < 8; ++nt) {
      v16h bf = fragF32(W_fc1 + (nt * 16 + lane15) * PRE + kc + kb);
      acc3[nt] = __builtin_amdgcn_wmma_f32_16x16x32_f16(
          false, af, false, bf, (short)0, acc3[nt], false, false);
    }
  }

  // ---- stage 4: logits = relu(queries) . W_keys + b_keys --------------------
  float wk[8], bq[8];
#pragma unroll
  for (int nt = 0; nt < 8; ++nt) {
    const int l = nt * 16 + lane15;
    wk[nt] = W_keys[l];
    bq[nt] = b_fc1[l];
  }
  float lg[8];
#pragma unroll
  for (int r = 0; r < 8; ++r) {
    float s = 0.f;
#pragma unroll
    for (int nt = 0; nt < 8; ++nt) {
      float q = fmaxf(acc3[nt][r] + bq[nt], 0.f);
      s += q * wk[nt];
    }
    lg[r] = s;
  }
#pragma unroll
  for (int o = 1; o <= 8; o <<= 1)
#pragma unroll
    for (int r = 0; r < 8; ++r) lg[r] += __shfl_xor(lg[r], o, 32);
  if (lane15 == 0) {
    const float bk = b_keys[0];
#pragma unroll
    for (int r = 0; r < 8; ++r) logits[b * NN + i0 + hi8 + r] = lg[r] + bk;
  }
}

// ---------------------------------------------------------------------------
// Kernel 4: softmax(logits/sqrt(2)) -> attention pooling -> output linear.
// One 1024-thread block per batch.
// ---------------------------------------------------------------------------
__global__ __launch_bounds__(1024) void k_pool(
    const float* __restrict__ logits, const _Float16* __restrict__ values,
    const float* __restrict__ W_out, const float* __restrict__ b_out,
    float* __restrict__ y) {
  __shared__ float sc[NN];
  __shared__ float red[32];
  __shared__ float part[16][64];
  __shared__ float vout[64];

  const int b = blockIdx.x, t = threadIdx.x;
  const int lane = t & 31, wid = t >> 5;

  float l = logits[b * NN + t] * 0.70710678118654752f;   // 1/sqrt(2)

  // max reduction
  float m = l;
#pragma unroll
  for (int o = 16; o; o >>= 1) m = fmaxf(m, __shfl_xor(m, o, 32));
  if (!lane) red[wid] = m;
  __syncthreads();
  if (t < 32) {
    float v = red[t];
#pragma unroll
    for (int o = 16; o; o >>= 1) v = fmaxf(v, __shfl_xor(v, o, 32));
    if (!t) red[0] = v;
  }
  __syncthreads();
  const float mx = red[0];
  float e = __expf(l - mx);
  sc[t] = e;
  __syncthreads();

  // sum reduction
  float s = e;
#pragma unroll
  for (int o = 16; o; o >>= 1) s += __shfl_xor(s, o, 32);
  if (!lane) red[wid] = s;
  __syncthreads();
  if (t < 32) {
    float v = red[t];
#pragma unroll
    for (int o = 16; o; o >>= 1) v += __shfl_xor(v, o, 32);
    if (!t) red[0] = v;
  }
  __syncthreads();
  const float inv = 1.f / red[0];

  // vect[p] = inv * sum_n e[n] * values[b][n][p]
  const int p = t & 63, g = t >> 6;           // 16 n-groups x 64 features
  const _Float16* vb = values + (size_t)b * NN * PRE;
  float acc = 0.f;
  for (int n = g; n < NN; n += 16)
    acc += sc[n] * (float)vb[n * PRE + p];
  part[g][p] = acc;
  __syncthreads();
#pragma unroll
  for (int st = 8; st; st >>= 1) {
    if (g < st) part[g][p] += part[g + st][p];
    __syncthreads();
  }
  if (t < 64) vout[t] = part[0][t] * inv * W_out[t];
  __syncthreads();
  if (t < 32) {
    float v = vout[t] + vout[t + 32];
#pragma unroll
    for (int o = 16; o; o >>= 1) v += __shfl_xor(v, o, 32);
    if (!t) y[b] = v + b_out[0];
  }
}

// ---------------------------------------------------------------------------
// Host-side launch wiring
// ---------------------------------------------------------------------------
extern "C" void kernel_launch(void* const* d_in, const int* in_sizes, int n_in,
                              void* d_out, int out_size, void* d_ws, size_t ws_size,
                              hipStream_t stream) {
  (void)in_sizes; (void)n_in; (void)out_size; (void)ws_size;
  const float* x      = (const float*)d_in[0];
  const float* A      = (const float*)d_in[1];
  const float* alpha  = (const float*)d_in[2];
  const float* W_pre  = (const float*)d_in[3];
  const float* b_pre  = (const float*)d_in[4];
  const float* W_fc1  = (const float*)d_in[5];
  const float* b_fc1  = (const float*)d_in[6];
  const float* W_keys = (const float*)d_in[7];
  const float* b_keys = (const float*)d_in[8];
  const float* W_out  = (const float*)d_in[9];
  const float* b_out  = (const float*)d_in[10];
  float* y = (float*)d_out;

  // workspace layout
  char* ws = (char*)d_ws;
  float*    disq   = (float*)ws;                                 // B*N f32   (128 KB)
  float*    logits = (float*)(ws + 131072);                      // B*N f32   (128 KB)
  _Float16* xsT    = (_Float16*)(ws + 262144);                   // B*F*N f16 (4 MB)
  _Float16* values = (_Float16*)(ws + 262144 + 4194304);         // B*N*PRE f16 (4 MB)

  k_degree<<<dim3((BB * NN) / 8), dim3(256), 0, stream>>>(A, disq);
  k_xsT  <<<dim3((BB * FF * NN) / 256), dim3(256), 0, stream>>>(x, disq, xsT);
  k_fused<<<dim3(BB * (NN / 64)), dim3(128), 0, stream>>>(
      A, x, alpha, disq, xsT, W_pre, b_pre, W_fc1, b_fc1, W_keys, b_keys,
      values, logits);
  k_pool <<<dim3(BB), dim3(1024), 0, stream>>>(logits, values, W_out, b_out, y);
}
